// FastGTNs_56453050138951
// MI455X (gfx1250) — matrix operand order repaired and checked by reading, not compiled, not run on hardware
//
#include <hip/hip_runtime.h>
#include <hip/hip_bf16.h>
#include <math.h>

// ---------------- problem constants (from reference) ----------------
#define N_NODES   100000
#define W_IN      256
#define W_OUTD    64
#define C_CH      2
#define T_ET      4
#define E_EDGES   800000
#define M_TGT     20000
#define NUM_CLS   4
#define BETA      0.5f

typedef __attribute__((ext_vector_type(16))) _Float16 v16h;
typedef __attribute__((ext_vector_type(8)))  _Float16 half8v;
typedef __attribute__((ext_vector_type(8)))  float    v8f;
typedef __attribute__((ext_vector_type(4)))  float    float4v;

// ============================================================================
// K1: prep — filt[c][t] = (1-beta)*softmax(conv_weight[c])[t]; zero the loss.
// ============================================================================
__global__ void prep_kernel(const float* __restrict__ cw,
                            float* __restrict__ filt,
                            float* __restrict__ out_loss) {
    if (threadIdx.x == 0) out_loss[0] = 0.0f;
    int c = (int)threadIdx.x;
    if (c < C_CH) {
        float mx = -1e30f;
        for (int t = 0; t < T_ET; ++t) mx = fmaxf(mx, cw[c * T_ET + t]);
        float e[T_ET]; float s = 0.0f;
        for (int t = 0; t < T_ET; ++t) { e[t] = __expf(cw[c * T_ET + t] - mx); s += e[t]; }
        float inv = (1.0f - BETA) / s;
        for (int t = 0; t < T_ET; ++t) filt[c * T_ET + t] = e[t] * inv;
    }
}

// ---- fragment builders (ISA §7.12.2 16-bit layouts) ------------------------
__device__ __forceinline__ v16h load_a_frag(const _Float16* sA, int row, int hf) {
    v16h a;
    #pragma unroll
    for (int v = 0; v < 8; ++v) {
        int K = ((v >> 2) << 4) + (hf << 3) + ((v & 3) << 1);
        const _Float16* p = &sA[row * 32 + K];
        a[2 * v] = p[0]; a[2 * v + 1] = p[1];
    }
    return a;
}
__device__ __forceinline__ v16h load_b_frag(const _Float16* colBase, int hf) {
    v16h b;
    #pragma unroll
    for (int v = 0; v < 8; ++v) {
        int K = ((v >> 2) << 4) + (hf << 3) + ((v & 3) << 1);
        b[2 * v] = colBase[K]; b[2 * v + 1] = colBase[K + 1];
    }
    return b;
}

// ============================================================================
// K2: GEMM1 — X_[c] = X @ Ws[c] via v_wmma_f32_16x16x32_f16.
//     Block: 64 rows x (2ch x 64 cols). 8 waves, each wave = 16x64 strip.
//     Double-buffered LDS A-tile, register-pipelined b128 staging.
//     Also writes Hbuf = BETA * X_ (fused init for the scatter stage).
// ============================================================================
__global__ __launch_bounds__(256)
void gemm1_wmma(const float* __restrict__ X, const float* __restrict__ Ws,
                float* __restrict__ Xc, float* __restrict__ Hbuf) {
    __shared__ _Float16 sW[C_CH * W_OUTD * W_IN];  // [c][col][k], 64 KB
    __shared__ _Float16 sA[2][64 * 32];            // double buffer, 2 x 4 KB

    const int tid = (int)threadIdx.x;

    // Preload Ws (f32, [c][k][col]) -> LDS f16 transposed [c][col][k].
    // float4 coalesced reads; 4 strided b16 stores each.
    #pragma unroll 4
    for (int i4 = tid; i4 < (C_CH * W_IN * W_OUTD) / 4; i4 += 256) {
        int i   = i4 * 4;
        int c   = i / (W_IN * W_OUTD);
        int r   = i % (W_IN * W_OUTD);
        int k   = r / W_OUTD;
        int col = r % W_OUTD;
        float4v w = *(const float4v*)&Ws[i];
        _Float16* dst = &sW[c * (W_OUTD * W_IN) + col * W_IN + k];
        dst[0 * W_IN] = (_Float16)w[0];
        dst[1 * W_IN] = (_Float16)w[1];
        dst[2 * W_IN] = (_Float16)w[2];
        dst[3 * W_IN] = (_Float16)w[3];
    }

    // Per-thread staging coords: 2048 elems / 256 thr = 8 (one half8 store).
    const int sr = tid >> 2;           // row 0..63
    const int sk = (tid & 3) * 8;      // k offset 0,8,16,24
    const int rowBase = (int)blockIdx.x * 64;
    int grow = rowBase + sr; if (grow >= N_NODES) grow = N_NODES - 1;
    const float* gX = &X[(size_t)grow * W_IN + sk];

    const int wave = tid >> 5, lane = tid & 31;
    const int ch = wave >> 2;                // 0..1
    const int rowOff = (wave & 3) * 16;      // 0,16,32,48
    const int mN = lane & 15;                // A row / B,C col within tile
    const int hf = lane >> 4;                // half-wave select

    v8f acc[4] = {};

    // Prime buffer 0 with chunk kb=0.
    {
        float4v r0 = *(const float4v*)(gX + 0);
        float4v r1 = *(const float4v*)(gX + 4);
        half8v h;
        #pragma unroll
        for (int j = 0; j < 4; ++j) { h[j] = (_Float16)r0[j]; h[4 + j] = (_Float16)r1[j]; }
        *(half8v*)&sA[0][sr * 32 + sk] = h;
    }
    __syncthreads();

    #pragma unroll
    for (int it = 0; it < W_IN / 32; ++it) {
        const int kb = it * 32;
        const int buf = it & 1;
        const bool more = (it + 1) < (W_IN / 32);

        // Issue next chunk's global loads early (overlap with WMMA).
        float4v n0, n1;
        if (more) {
            n0 = *(const float4v*)(gX + kb + 32);
            n1 = *(const float4v*)(gX + kb + 36);
        }

        // Compute on current buffer.
        v16h a = load_a_frag(sA[buf], rowOff + mN, hf);
        #pragma unroll
        for (int nt = 0; nt < 4; ++nt) {
            int col = nt * 16 + mN;
            v16h b = load_b_frag(&sW[ch * (W_OUTD * W_IN) + col * W_IN + kb], hf);
            acc[nt] = __builtin_amdgcn_wmma_f32_16x16x32_f16(
                false, a, false, b, (short)0, acc[nt], false, false);
        }

        // Stage next chunk into the other buffer; one barrier per step.
        if (more) {
            half8v h;
            #pragma unroll
            for (int j = 0; j < 4; ++j) { h[j] = (_Float16)n0[j]; h[4 + j] = (_Float16)n1[j]; }
            *(half8v*)&sA[buf ^ 1][sr * 32 + sk] = h;
            __syncthreads();
        }
    }

    // Store D (C/D layout: VGPR v -> row v + 8*hf) + fused Hbuf = BETA * X_.
    #pragma unroll
    for (int nt = 0; nt < 4; ++nt) {
        #pragma unroll
        for (int v = 0; v < 8; ++v) {
            int row = rowBase + rowOff + v + 8 * hf;
            int col = nt * 16 + mN;
            if (row < N_NODES) {
                float d = acc[nt][v];
                size_t off = (size_t)ch * N_NODES * W_OUTD + (size_t)row * W_OUTD + col;
                Xc[off]   = d;
                Hbuf[off] = BETA * d;
            }
        }
    }
}

// ============================================================================
// K4: SpMM scatter — Hbuf[c][row] += (ev*filt)[edge] * X_[c][colnode].
//     64 threads per edge (one dim each), coalesced 256B gather, f32 atomics.
// ============================================================================
__global__ __launch_bounds__(256)
void spmm_scatter(const float* __restrict__ ev, const int* __restrict__ ei,
                  const float* __restrict__ filt, const float* __restrict__ Xc,
                  float* __restrict__ Hbuf) {
    const long long idx = (long long)blockIdx.x * 4 + threadIdx.y;
    const long long per_ch = (long long)T_ET * E_EDGES;
    if (idx >= (long long)C_CH * per_ch) return;
    const int c = (int)(idx / per_ch);
    const long long j = idx % per_ch;
    const int t = (int)(j / E_EDGES);
    const int e = (int)(j % E_EDGES);

    const float val = ev[(size_t)t * E_EDGES + e] * filt[c * T_ET + t];
    const int row = ei[((size_t)t * 2 + 0) * E_EDGES + e];
    const int cn  = ei[((size_t)t * 2 + 1) * E_EDGES + e];
    const int d = (int)threadIdx.x;  // 0..63

    const size_t cbase = (size_t)c * N_NODES * W_OUTD;
    const float msg = val * Xc[cbase + (size_t)cn * W_OUTD + d];
    __hip_atomic_fetch_add(&Hbuf[cbase + (size_t)row * W_OUTD + d], msg,
                           __ATOMIC_RELAXED, __HIP_MEMORY_SCOPE_AGENT);
}

// ============================================================================
// K6: GEMM2 — H2 = relu( relu(Hbuf)_cat[N,128] @ linear1_W.T + b1 ).
//     ReLU + concat + f32->f16 fused into the staged A-tile. Double-buffered.
//     Block: 128 rows x 64 cols; 8 waves, each a 16-row strip.
// ============================================================================
__global__ __launch_bounds__(256)
void gemm2_wmma(const float* __restrict__ Hbuf, const float* __restrict__ W1,
                const float* __restrict__ b1, float* __restrict__ H2) {
    __shared__ _Float16 sW[W_OUTD * 128];   // [o][k] straight copy, 16 KB
    __shared__ _Float16 sA[2][128 * 32];    // double buffer, 2 x 8 KB

    const int tid = (int)threadIdx.x;

    // linear1_W f32 [o][k] -> LDS f16 same layout; k-pairs contiguous.
    #pragma unroll 4
    for (int i4 = tid; i4 < (W_OUTD * 128) / 4; i4 += 256) {
        float4v w = *(const float4v*)&W1[i4 * 4];
        half8v* unused = nullptr; (void)unused;
        _Float16* dst = &sW[i4 * 4];
        dst[0] = (_Float16)w[0]; dst[1] = (_Float16)w[1];
        dst[2] = (_Float16)w[2]; dst[3] = (_Float16)w[3];
    }

    // Per-thread staging coords: 4096 elems / 256 thr = 16 (two half8 stores).
    const int sr = tid >> 1;           // row 0..127
    const int sk = (tid & 1) * 16;     // k offset 0 or 16
    const int rowBase = (int)blockIdx.x * 128;
    int grow = rowBase + sr; if (grow >= N_NODES) grow = N_NODES - 1;

    const int wave = tid >> 5, lane = tid & 31;
    const int rowOff = wave * 16;
    const int mN = lane & 15, hf = lane >> 4;

    v8f acc[4] = {};

    // Stage helper: chunk kb covers k = kb..kb+31, all within one channel.
    auto stage = [&](int kb, int buf) {
        const int c = kb >> 6;                 // channel plane
        const int ob = (kb & 63) + sk;         // feature offset within plane
        const float* src = &Hbuf[(size_t)c * N_NODES * W_OUTD +
                                 (size_t)grow * W_OUTD + ob];
        float4v r0 = *(const float4v*)(src + 0);
        float4v r1 = *(const float4v*)(src + 4);
        float4v r2 = *(const float4v*)(src + 8);
        float4v r3 = *(const float4v*)(src + 12);
        half8v h0, h1;
        #pragma unroll
        for (int j = 0; j < 4; ++j) {
            h0[j]     = (_Float16)fmaxf(r0[j], 0.0f);
            h0[4 + j] = (_Float16)fmaxf(r1[j], 0.0f);
            h1[j]     = (_Float16)fmaxf(r2[j], 0.0f);
            h1[4 + j] = (_Float16)fmaxf(r3[j], 0.0f);
        }
        *(half8v*)&sA[buf][sr * 32 + sk + 0] = h0;
        *(half8v*)&sA[buf][sr * 32 + sk + 8] = h1;
    };

    stage(0, 0);
    __syncthreads();

    #pragma unroll
    for (int it = 0; it < 4; ++it) {
        const int kb = it * 32;
        const int buf = it & 1;
        const bool more = it < 3;

        v16h a = load_a_frag(sA[buf], rowOff + mN, hf);
        #pragma unroll
        for (int nt = 0; nt < 4; ++nt) {
            int col = nt * 16 + mN;
            v16h b = load_b_frag(&sW[col * 128 + kb], hf);
            acc[nt] = __builtin_amdgcn_wmma_f32_16x16x32_f16(
                false, a, false, b, (short)0, acc[nt], false, false);
        }

        if (more) {
            stage(kb + 32, buf ^ 1);
            __syncthreads();
        }
    }

    #pragma unroll
    for (int nt = 0; nt < 4; ++nt) {
        #pragma unroll
        for (int v = 0; v < 8; ++v) {
            int row = rowBase + rowOff + v + 8 * hf;
            int col = nt * 16 + mN;
            if (row < N_NODES) {
                float d = acc[nt][v] + b1[col];
                H2[(size_t)row * W_OUTD + col] = d > 0.0f ? d : 0.0f;
            }
        }
    }
}

// ============================================================================
// K7: head — y = H2[target_x] @ lin_W.T + lin_b; log-softmax NLL mean.
// ============================================================================
__global__ __launch_bounds__(256)
void head_kernel(const float* __restrict__ H2, const float* __restrict__ lW,
                 const float* __restrict__ lb, const int* __restrict__ tx,
                 const int* __restrict__ tg, float* __restrict__ out) {
    int m = (int)(blockIdx.x * blockDim.x + threadIdx.x);
    if (m >= M_TGT) return;
    const int row = tx[m];
    const float* h = &H2[(size_t)row * W_OUTD];

    float y[NUM_CLS];
    #pragma unroll
    for (int k = 0; k < NUM_CLS; ++k) {
        float s = lb[k];
        for (int o = 0; o < W_OUTD; ++o) s = fmaf(h[o], lW[k * W_OUTD + o], s);
        y[k] = s;
    }
    float mx = y[0];
    #pragma unroll
    for (int k = 1; k < NUM_CLS; ++k) mx = fmaxf(mx, y[k]);
    float se = 0.0f;
    #pragma unroll
    for (int k = 0; k < NUM_CLS; ++k) se += __expf(y[k] - mx);
    const float lse = mx + __logf(se);

    const float contrib = -(y[tg[m]] - lse) * (1.0f / (float)M_TGT);
    __hip_atomic_fetch_add(out, contrib, __ATOMIC_RELAXED, __HIP_MEMORY_SCOPE_AGENT);

    #pragma unroll
    for (int k = 0; k < NUM_CLS; ++k) out[1 + (size_t)m * NUM_CLS + k] = y[k];
}

// ============================================================================
extern "C" void kernel_launch(void* const* d_in, const int* in_sizes, int n_in,
                              void* d_out, int out_size, void* d_ws, size_t ws_size,
                              hipStream_t stream) {
    const float* X  = (const float*)d_in[0];
    const float* ev = (const float*)d_in[1];
    const float* Ws = (const float*)d_in[2];
    const float* cw = (const float*)d_in[3];
    const float* W1 = (const float*)d_in[4];
    const float* b1 = (const float*)d_in[5];
    const float* lW = (const float*)d_in[6];
    const float* lb = (const float*)d_in[7];
    const int*   ei = (const int*)d_in[8];
    const int*   tx = (const int*)d_in[9];
    const int*   tg = (const int*)d_in[10];
    float* out = (float*)d_out;

    // Workspace layout (256B-aligned): filt | X_ | Hbuf | H2  (~128 MB total)
    char* ws = (char*)d_ws;
    const size_t CNO = (size_t)C_CH * N_NODES * W_OUTD * sizeof(float);  // 51.2 MB
    float* filt = (float*)ws;
    float* Xc   = (float*)(ws + 256);
    float* Hbuf = (float*)(ws + 256 + CNO);
    float* H2   = (float*)(ws + 256 + 2 * CNO);

    const int g1 = (N_NODES + 63) / 64;                                   // 1563
    const long long edges = (long long)C_CH * T_ET * E_EDGES;             // 6.4M
    const int g2 = (int)((edges + 3) / 4);                                // 1.6M
    const int g3 = (N_NODES + 127) / 128;                                 // 782
    const int g4 = (M_TGT + 255) / 256;                                   // 79

    prep_kernel <<<1, 32, 0, stream>>>(cw, filt, out);
    gemm1_wmma  <<<g1, 256, 0, stream>>>(X, Ws, Xc, Hbuf);
    spmm_scatter<<<g2, dim3(64, 4), 0, stream>>>(ev, ei, filt, Xc, Hbuf);
    gemm2_wmma  <<<g3, 256, 0, stream>>>(Hbuf, W1, b1, H2);
    head_kernel <<<g4, 256, 0, stream>>>(H2, lW, lb, tx, tg, out);
}
